// GNN_Transformer_8512625180878
// MI455X (gfx1250) — matrix operand order repaired and checked by reading, not compiled, not run on hardware
//
#include <hip/hip_runtime.h>
#include <hip/hip_bf16.h>

typedef _Float16 f16;
typedef __attribute__((ext_vector_type(16))) _Float16 v16h;
typedef __attribute__((ext_vector_type(8)))  _Float16 v8h;
typedef __attribute__((ext_vector_type(8)))  float    v8f;

// ---------------------------------------------------------------------------
// WMMA fragment helpers (gfx1250 v_wmma_f32_16x16x32_f16 layouts)
// A 16x32 (MxK): lane l holds row m=l&15; halfs 0..7 -> K = ((l>=16)?8:0)+h,
//                halfs 8..15 -> K = 16 + ((l>=16)?8:0) + (h-8)   => two b128 loads
// B 32x16 (KxN) supplied transposed as Bt[N][K]: lane l holds col n=l&15,
//                same K mapping as A (halfs 0..7 -> 8*(l>=16)+h, 8..15 -> +16) => two b128 loads
// C/D 16x16 f32: vgpr r, lane l -> (m = r + 8*(l>=16), n = l&15)
// ---------------------------------------------------------------------------
static __device__ __forceinline__ v16h cat16(v8h lo, v8h hi){
  v16h r;
#pragma unroll
  for (int i = 0; i < 8; ++i){ r[i] = lo[i]; r[i+8] = hi[i]; }
  return r;
}

static __device__ __forceinline__ v16h fragA(const f16* base, int lda, int lane){
  const int m  = lane & 15;
  const int kb = (lane >> 4) << 3;          // 0 or 8
  v8h lo = *(const v8h*)(base + m * lda + kb);
  v8h hi = *(const v8h*)(base + m * lda + kb + 16);
  return cat16(lo, hi);
}

static __device__ __forceinline__ v16h fragB(const f16* base, int ldb, int lane){
  const int n  = lane & 15;
  const int kb = (lane >> 4) << 3;          // 0 or 8
  v8h lo = *(const v8h*)(base + n * ldb + kb);
  v8h hi = *(const v8h*)(base + n * ldb + kb + 16);
  return cat16(lo, hi);
}

static __device__ __forceinline__ v8f wmma_f16(v16h a, v16h b, v8f c){
  return __builtin_amdgcn_wmma_f32_16x16x32_f16(false, a, false, b, (short)0, c, false, false);
}

// ---------------------------------------------------------------------------
// Weight prep: f32 W[K][N] -> f16 Wt[Npad][Kpad] (transposed, zero padded)
// ---------------------------------------------------------------------------
__global__ void prep_wt(const float* __restrict__ W, f16* __restrict__ Wt,
                        int K, int N, int Kpad, int Npad){
  int idx = blockIdx.x * blockDim.x + threadIdx.x;
  int total = Kpad * Npad;
  if (idx >= total) return;
  int n = idx / Kpad;
  int k = idx - n * Kpad;
  Wt[idx] = (k < K && n < N) ? (f16)W[(size_t)k * N + n] : (f16)0.f;
}

// ---------------------------------------------------------------------------
// Fused GAT1 (VALU, K=4) + GATv2* + GAT2 (WMMA, K=256) + GATv2* + node-mean
// One block = NG groups of 16 nodes: each weight B-fragment is reused for
// NG A-tiles (halves L2 weight traffic vs 1 group/block).
// Output: pooled[g][256] (f32)
// ---------------------------------------------------------------------------
static constexpr int NG = 2;   // groups per block

static __device__ __forceinline__ void gat_scores_h(const f16* xl, const f16* xr,
                                                    const float* __restrict__ att,
                                                    float* wh, float* wsf, int tid){
  if (tid < NG * 30){
    const int gl = tid / 30;
    const int t  = tid - gl * 30;
    const int n  = 1 + (t % 15);
    const int h  = t / 15;
    const f16* hubp = xl + gl * 4096 + h * 128;           // node 0
    const f16* qp   = xr + gl * 4096 + n * 256 + h * 128;
    const f16* sp   = xl + gl * 4096 + n * 256 + h * 128;
    const float* ap = att + h * 128;
    float e1 = 0.f, e2 = 0.f;
    for (int c = 0; c < 128; ++c){
      float a = ap[c], q = (float)qp[c];
      float v1 = (float)hubp[c] + q; v1 = v1 > 0.f ? v1 : 0.2f * v1;
      float v2 = (float)sp[c]   + q; v2 = v2 > 0.f ? v2 : 0.2f * v2;
      e1 += a * v1; e2 += a * v2;
    }
    float m  = fmaxf(e1, e2);
    float a1 = __expf(e1 - m), a2 = __expf(e2 - m);
    float inv = 1.f / (a1 + a2);
    wh [gl * 32 + n * 2 + h] = a1 * inv;
    wsf[gl * 32 + n * 2 + h] = a2 * inv;
  }
}

__global__ __launch_bounds__(256)
void gat_fused_kernel(const float* __restrict__ x,
                      const float* __restrict__ Wl1, const float* __restrict__ bl1,
                      const float* __restrict__ Wr1, const float* __restrict__ br1,
                      const float* __restrict__ att1, const float* __restrict__ bias1,
                      const f16*  __restrict__ Wl2t, const float* __restrict__ bl2,
                      const f16*  __restrict__ Wr2t, const float* __restrict__ br2,
                      const float* __restrict__ att2, const float* __restrict__ bias2,
                      float* __restrict__ pooled){
  __shared__ f16   sXLh[NG * 16 * 256];
  __shared__ f16   sXRh[NG * 16 * 256];
  __shared__ f16   sAh [NG * 16 * 256];
  __shared__ float sX[NG * 64];
  __shared__ float sWh[NG * 32], sWs[NG * 32];

  const int g0  = blockIdx.x * NG;
  const int tid = threadIdx.x;

  if (tid < NG * 64) sX[tid] = x[(size_t)g0 * 64 + tid];
  __syncthreads();

  // GAT1 linear: xl/xr = x @ W + b  (K=4) for all NG groups
  {
    const int j = tid;
    const float l0 = Wl1[j], l1 = Wl1[256 + j], l2 = Wl1[512 + j], l3 = Wl1[768 + j];
    const float r0 = Wr1[j], r1 = Wr1[256 + j], r2 = Wr1[512 + j], r3 = Wr1[768 + j];
    const float bl = bl1[j], br = br1[j];
#pragma unroll
    for (int gl = 0; gl < NG; ++gl){
      for (int n = 0; n < 16; ++n){
        const float x0 = sX[gl*64 + n*4],     x1 = sX[gl*64 + n*4 + 1];
        const float x2 = sX[gl*64 + n*4 + 2], x3 = sX[gl*64 + n*4 + 3];
        sXLh[gl*4096 + n*256 + j] =
            (f16)(fmaf(x0, l0, fmaf(x1, l1, fmaf(x2, l2, x3 * l3))) + bl);
        sXRh[gl*4096 + n*256 + j] =
            (f16)(fmaf(x0, r0, fmaf(x1, r1, fmaf(x2, r2, x3 * r3))) + br);
      }
    }
  }
  __syncthreads();

  gat_scores_h(sXLh, sXRh, att1, sWh, sWs, tid);
  __syncthreads();

  // combine + bias + leaky(0.01) -> f16 A tiles for GAT2
  {
    const int j = tid, h = j >> 7;
    const float b = bias1[j];
#pragma unroll
    for (int gl = 0; gl < NG; ++gl){
      const float hub = (float)sXLh[gl*4096 + j];
      float v0 = hub + b; v0 = v0 > 0.f ? v0 : 0.01f * v0;
      sAh[gl*4096 + j] = (f16)v0;
      for (int n = 1; n < 16; ++n){
        float v = sWh[gl*32 + n*2 + h] * hub
                + sWs[gl*32 + n*2 + h] * (float)sXLh[gl*4096 + n*256 + j] + b;
        v = v > 0.f ? v : 0.01f * v;
        sAh[gl*4096 + n*256 + j] = (f16)v;
      }
    }
  }
  __syncthreads();

  // GAT2 linear via WMMA: 32 tile-jobs (16 for xl, 16 for xr), K=256 (8 steps)
  // Each B fragment load feeds NG WMMAs (one per group).
  {
    const int wave = tid >> 5, lane = tid & 31;
    for (int job = wave; job < 32; job += 8){
      const int which = job >> 4;
      const int nt    = job & 15;
      const f16*   Bt = which ? Wr2t : Wl2t;
      const float* bs = which ? br2  : bl2;
      f16*         Xo = which ? sXRh : sXLh;
      const int col = (nt << 4) + (lane & 15);
      const float bi = bs[col];
      v8f c[NG];
#pragma unroll
      for (int gl = 0; gl < NG; ++gl)
#pragma unroll
        for (int i = 0; i < 8; ++i) c[gl][i] = bi;
      const f16* Bbase = Bt + (size_t)(nt << 4) * 256;
      for (int kk = 0; kk < 8; ++kk){
        if (kk + 1 < 8) __builtin_prefetch(Bbase + (kk + 1) * 32, 0, 3);
        v16h b = fragB(Bbase + kk * 32, 256, lane);
#pragma unroll
        for (int gl = 0; gl < NG; ++gl){
          v16h a = fragA(sAh + gl*4096 + kk * 32, 256, lane);
          c[gl] = wmma_f16(a, b, c[gl]);
        }
      }
#pragma unroll
      for (int gl = 0; gl < NG; ++gl){
#pragma unroll
        for (int r = 0; r < 8; ++r){
          const int m = r + ((lane >> 4) << 3);
          Xo[gl*4096 + m * 256 + col] = (f16)c[gl][r];
        }
      }
    }
  }
  __syncthreads();

  gat_scores_h(sXLh, sXRh, att2, sWh, sWs, tid);
  __syncthreads();

  // combine + bias + leaky(0.01) + mean over 16 nodes -> pooled
  {
    const int j = tid, h = j >> 7;
    const float b = bias2[j];
#pragma unroll
    for (int gl = 0; gl < NG; ++gl){
      const float hub = (float)sXLh[gl*4096 + j];
      float v0 = hub + b; v0 = v0 > 0.f ? v0 : 0.01f * v0;
      float acc = v0;
      for (int n = 1; n < 16; ++n){
        float v = sWh[gl*32 + n*2 + h] * hub
                + sWs[gl*32 + n*2 + h] * (float)sXLh[gl*4096 + n*256 + j] + b;
        v = v > 0.f ? v : 0.01f * v;
        acc += v;
      }
      pooled[(size_t)(g0 + gl) * 256 + j] = acc * (1.f / 16.f);
    }
  }
}

// ---------------------------------------------------------------------------
// Build concat(pooled, x_scalar) as f16 [M][288] (zero padded K 260->288)
// ---------------------------------------------------------------------------
__global__ void build_tin_kernel(const float* __restrict__ pooled,
                                 const float* __restrict__ xs,
                                 f16* __restrict__ A, int M){
  int idx = blockIdx.x * blockDim.x + threadIdx.x;
  if (idx >= M * 288) return;
  int row = idx / 288, k = idx - row * 288;
  float v = (k < 256) ? pooled[(size_t)row * 256 + k]
                      : ((k < 260) ? xs[(size_t)row * 4 + (k - 256)] : 0.f);
  A[idx] = (f16)v;
}

// ---------------------------------------------------------------------------
// Generic WMMA GEMM: C[M][N] = A(f16 [M][Kp]) @ Bt(f16 [Npad][Kp])^T + bias
// Paired N-tiles per job: one A fragment feeds two WMMAs (halves A traffic).
// optional relu; writes f32 out (ld=N) and/or f16 out (ld=Npad, pad zeroed)
// blockDim 256 (8 waves), 64 rows per block; Ntiles must be even.
// ---------------------------------------------------------------------------
__global__ __launch_bounds__(256)
void gemm_wmma_kernel(const f16* __restrict__ A, const f16* __restrict__ Bt,
                      const float* __restrict__ bias,
                      float* __restrict__ Cf, f16* __restrict__ Ch,
                      int N, int Ntiles, int Kp, int relu){
  const int row0 = blockIdx.x * 64;
  const int wave = threadIdx.x >> 5;
  const int lane = threadIdx.x & 31;
  const int Ksteps = Kp >> 5;
  const int Npad   = Ntiles << 4;
  const int half   = Ntiles >> 1;
  const int jobs   = half << 2;          // 4 M-tiles x (Ntiles/2) pairs
  for (int job = wave; job < jobs; job += 8){
    const int mt  = job / half;
    const int np  = job - mt * half;
    const int nt0 = np << 1;
    const int col0 = (nt0 << 4) + (lane & 15);
    const int col1 = col0 + 16;
    const float bi0 = (bias != nullptr && col0 < N) ? bias[col0] : 0.f;
    const float bi1 = (bias != nullptr && col1 < N) ? bias[col1] : 0.f;
    v8f c0, c1;
#pragma unroll
    for (int i = 0; i < 8; ++i){ c0[i] = bi0; c1[i] = bi1; }
    const f16* Abase = A  + (size_t)(row0 + (mt << 4)) * Kp;
    const f16* Bb0   = Bt + (size_t)(nt0 << 4) * Kp;
    const f16* Bb1   = Bb0 + (size_t)16 * Kp;
    for (int kk = 0; kk < Ksteps; ++kk){
      if (kk + 1 < Ksteps){
        __builtin_prefetch(Bb0 + (kk + 1) * 32, 0, 3);
        __builtin_prefetch(Bb1 + (kk + 1) * 32, 0, 3);
      }
      v16h a  = fragA(Abase + kk * 32, Kp, lane);
      v16h b0 = fragB(Bb0 + kk * 32, Kp, lane);
      v16h b1 = fragB(Bb1 + kk * 32, Kp, lane);
      c0 = wmma_f16(a, b0, c0);
      c1 = wmma_f16(a, b1, c1);
    }
#pragma unroll
    for (int r = 0; r < 8; ++r){
      float v0 = c0[r], v1 = c1[r];
      if (relu){ v0 = fmaxf(v0, 0.f); v1 = fmaxf(v1, 0.f); }
      const int m = row0 + (mt << 4) + r + ((lane >> 4) << 3);
      if (Cf != nullptr){
        if (col0 < N) Cf[(size_t)m * N + col0] = v0;
        if (col1 < N) Cf[(size_t)m * N + col1] = v1;
      }
      if (Ch != nullptr){
        Ch[(size_t)m * Npad + col0] = (col0 < N) ? (f16)v0 : (f16)0.f;
        Ch[(size_t)m * Npad + col1] = (col1 < N) ? (f16)v1 : (f16)0.f;
      }
    }
  }
}

// ---------------------------------------------------------------------------
// Positional encoding add (in-place on f32 t) + emit padded f16 copy
// ---------------------------------------------------------------------------
__global__ void pe_kernel(float* __restrict__ t, f16* __restrict__ th, int M){
  int idx = blockIdx.x * blockDim.x + threadIdx.x;
  if (idx >= M * 128) return;
  int row = idx >> 7, col = idx & 127;
  if (col < 124){
    int s  = row & 63;
    int i2 = col & ~1;
    float freq = __expf((float)i2 * (-9.210340371976184f / 124.f));
    float ang  = (float)s * freq;
    float pe   = (col & 1) ? __cosf(ang) : __sinf(ang);
    float v = t[(size_t)row * 124 + col] + pe;
    t[(size_t)row * 124 + col] = v;
    th[idx] = (f16)v;
  } else {
    th[idx] = (f16)0.f;
  }
}

// ---------------------------------------------------------------------------
// Per-(batch,head) attention: scores WMMA (K=32) -> softmax -> context WMMA
// ---------------------------------------------------------------------------
__global__ __launch_bounds__(128)
void attn_kernel(const f16* __restrict__ q, const f16* __restrict__ k,
                 const f16* __restrict__ v, f16* __restrict__ o){
  __shared__ f16   sQ[64 * 32];
  __shared__ f16   sK[64 * 32];
  __shared__ f16   sVT[32 * 64];
  __shared__ float sS[64 * 64];
  __shared__ f16   sA[64 * 64];

  const int b = blockIdx.x >> 2, h = blockIdx.x & 3;
  const int tid = threadIdx.x;
  const size_t base = (size_t)b * 64;

  for (int idx = tid; idx < 2048; idx += 128){
    int s = idx >> 5, d = idx & 31;
    f16 qv = (d < 31) ? q[(base + s) * 128 + h * 31 + d] : (f16)0.f;
    f16 kv = (d < 31) ? k[(base + s) * 128 + h * 31 + d] : (f16)0.f;
    f16 vv = (d < 31) ? v[(base + s) * 128 + h * 31 + d] : (f16)0.f;
    sQ[idx] = qv; sK[idx] = kv; sVT[d * 64 + s] = vv;
  }
  __syncthreads();

  const int wave = tid >> 5, lane = tid & 31;
  const float scale = 0.1796053020267749f;   // 1/sqrt(31)

  for (int job = wave; job < 16; job += 4){
    int mt = job >> 2, nt = job & 3;
    v16h a = fragA(sQ + (size_t)(mt << 4) * 32, 32, lane);
    v16h bb = fragB(sK + (size_t)(nt << 4) * 32, 32, lane);
    v8f c;
#pragma unroll
    for (int i = 0; i < 8; ++i) c[i] = 0.f;
    c = wmma_f16(a, bb, c);
#pragma unroll
    for (int r = 0; r < 8; ++r){
      int m = (mt << 4) + r + ((lane >> 4) << 3);
      int n = (nt << 4) + (lane & 15);
      sS[m * 64 + n] = c[r] * scale;
    }
  }
  __syncthreads();

  if (tid < 64){
    float mx = -3.4e38f;
    for (int j = 0; j < 64; ++j) mx = fmaxf(mx, sS[tid * 64 + j]);
    float sum = 0.f;
    for (int j = 0; j < 64; ++j){
      float e = __expf(sS[tid * 64 + j] - mx);
      sS[tid * 64 + j] = e; sum += e;
    }
    float inv = 1.f / sum;
    for (int j = 0; j < 64; ++j) sA[tid * 64 + j] = (f16)(sS[tid * 64 + j] * inv);
  }
  __syncthreads();

  for (int job = wave; job < 8; job += 4){
    int mt = job >> 1, nt = job & 1;
    v8f c;
#pragma unroll
    for (int i = 0; i < 8; ++i) c[i] = 0.f;
    for (int kk = 0; kk < 2; ++kk){
      v16h a = fragA(sA  + (size_t)(mt << 4) * 64 + kk * 32, 64, lane);
      v16h bb = fragB(sVT + (size_t)(nt << 4) * 64 + kk * 32, 64, lane);
      c = wmma_f16(a, bb, c);
    }
#pragma unroll
    for (int r = 0; r < 8; ++r){
      int s  = (mt << 4) + r + ((lane >> 4) << 3);
      int cl = (nt << 4) + (lane & 15);
      if (cl < 31) o[(base + s) * 128 + h * 31 + cl] = (f16)c[r];
    }
  }
  // zero pad cols 124..127 of the f16 output (written once, by head 3 block)
  if (h == 3){
    for (int idx = tid; idx < 256; idx += 128){
      int s = idx >> 2;
      o[(base + s) * 128 + 124 + (idx & 3)] = (f16)0.f;
    }
  }
}

// ---------------------------------------------------------------------------
// Residual + LayerNorm(124): outf = LN(a + b); also emit padded f16 copy
// ---------------------------------------------------------------------------
__global__ void ln_kernel(const float* __restrict__ a, const float* __restrict__ bsrc,
                          const float* __restrict__ g, const float* __restrict__ be,
                          float* __restrict__ outf, f16* __restrict__ outh, int M){
  int row = blockIdx.x * blockDim.x + threadIdx.x;
  if (row >= M) return;
  const float* pa = a    + (size_t)row * 124;
  const float* pb = bsrc + (size_t)row * 124;
  float mu = 0.f;
  for (int i = 0; i < 124; ++i) mu += pa[i] + pb[i];
  mu *= (1.f / 124.f);
  float var = 0.f;
  for (int i = 0; i < 124; ++i){ float d = pa[i] + pb[i] - mu; var += d * d; }
  var *= (1.f / 124.f);
  float inv = rsqrtf(var + 1e-5f);
  for (int i = 0; i < 128; ++i){
    if (i < 124){
      float v = (pa[i] + pb[i] - mu) * inv * g[i] + be[i];
      outf[(size_t)row * 124 + i] = v;
      outh[(size_t)row * 128 + i] = (f16)v;
    } else {
      outh[(size_t)row * 128 + i] = (f16)0.f;
    }
  }
}

// ---------------------------------------------------------------------------
// Final: mean over S=64 then @ W_out(124x6)+b_out
// ---------------------------------------------------------------------------
__global__ __launch_bounds__(128)
void final_kernel(const float* __restrict__ t, const float* __restrict__ Wout,
                  const float* __restrict__ bout, float* __restrict__ out){
  __shared__ float mean[124];
  const int b = blockIdx.x, tid = threadIdx.x;
  if (tid < 124){
    float s = 0.f;
    for (int r = 0; r < 64; ++r) s += t[((size_t)b * 64 + r) * 124 + tid];
    mean[tid] = s * (1.f / 64.f);
  }
  __syncthreads();
  if (tid < 6){
    float s = bout[tid];
    for (int c = 0; c < 124; ++c) s += mean[c] * Wout[c * 6 + tid];
    out[b * 6 + tid] = s;
  }
}

// ---------------------------------------------------------------------------
// Host
// ---------------------------------------------------------------------------
extern "C" void kernel_launch(void* const* d_in, const int* in_sizes, int n_in,
                              void* d_out, int out_size, void* d_ws, size_t ws_size,
                              hipStream_t stream){
  (void)in_sizes; (void)n_in; (void)out_size; (void)ws_size;
  const int G = 16384;   // B*S groups / token rows
  const int M = 16384;

  // inputs in setup_inputs() dict insertion order
  const float* X    = (const float*)d_in[0];
  const float* XS   = (const float*)d_in[1];
  const float* g1Wl = (const float*)d_in[2];
  const float* g1bl = (const float*)d_in[3];
  const float* g1Wr = (const float*)d_in[4];
  const float* g1br = (const float*)d_in[5];
  const float* g1at = (const float*)d_in[6];
  const float* g1bi = (const float*)d_in[7];
  const float* g2Wl = (const float*)d_in[8];
  const float* g2bl = (const float*)d_in[9];
  const float* g2Wr = (const float*)d_in[10];
  const float* g2br = (const float*)d_in[11];
  const float* g2at = (const float*)d_in[12];
  const float* g2bi = (const float*)d_in[13];
  const float* Win  = (const float*)d_in[14];
  const float* bin  = (const float*)d_in[15];
  const float* Wout = (const float*)d_in[48];
  const float* bout = (const float*)d_in[49];

  char* ws = (char*)d_ws;
  size_t off = 0;
  auto alloc = [&](size_t bytes)->char*{
    char* p = ws + off;
    off += (bytes + 255) & ~(size_t)255;
    return p;
  };

  f16* Wl2t = (f16*)alloc(256 * 256 * 2);
  f16* Wr2t = (f16*)alloc(256 * 256 * 2);
  f16* Wint = (f16*)alloc(128 * 288 * 2);
  f16 *Wqt[2], *Wkt[2], *Wvt[2], *Wot[2], *W1t[2], *W2t[2];
  for (int l = 0; l < 2; ++l){
    Wqt[l] = (f16*)alloc(128 * 128 * 2);
    Wkt[l] = (f16*)alloc(128 * 128 * 2);
    Wvt[l] = (f16*)alloc(128 * 128 * 2);
    Wot[l] = (f16*)alloc(128 * 128 * 2);
    W1t[l] = (f16*)alloc(64  * 128 * 2);
    W2t[l] = (f16*)alloc(128 * 64  * 2);
  }
  float* pooled = (float*)alloc((size_t)G * 256 * 4);
  f16*   Ain    = (f16*)  alloc((size_t)M * 288 * 2);
  float* t      = (float*)alloc((size_t)M * 124 * 4);
  float* t2     = (float*)alloc((size_t)M * 124 * 4);
  float* gbuf   = (float*)alloc((size_t)M * 124 * 4);
  f16*   th     = (f16*)  alloc((size_t)M * 128 * 2);
  f16*   t2h    = (f16*)  alloc((size_t)M * 128 * 2);
  f16*   qh     = (f16*)  alloc((size_t)M * 128 * 2);
  f16*   kh     = (f16*)  alloc((size_t)M * 128 * 2);
  f16*   vh     = (f16*)  alloc((size_t)M * 128 * 2);
  f16*   oh     = (f16*)  alloc((size_t)M * 128 * 2);
  f16*   ffh    = (f16*)  alloc((size_t)M * 64  * 2);

  auto prep = [&](const float* W, f16* Wt, int K, int N, int Kpad, int Npad){
    int total = Kpad * Npad;
    prep_wt<<<(total + 255) / 256, 256, 0, stream>>>(W, Wt, K, N, Kpad, Npad);
  };

  prep(g2Wl, Wl2t, 256, 256, 256, 256);
  prep(g2Wr, Wr2t, 256, 256, 256, 256);
  prep(Win,  Wint, 260, 124, 288, 128);
  for (int l = 0; l < 2; ++l){
    const int base = 16 + 16 * l;
    prep((const float*)d_in[base + 0], Wqt[l], 124, 124, 128, 128);
    prep((const float*)d_in[base + 2], Wkt[l], 124, 124, 128, 128);
    prep((const float*)d_in[base + 4], Wvt[l], 124, 124, 128, 128);
    prep((const float*)d_in[base + 6], Wot[l], 124, 124, 128, 128);
    prep((const float*)d_in[base + 8], W1t[l], 124, 64,  128, 64);
    prep((const float*)d_in[base +10], W2t[l], 64,  124, 64,  128);
  }

  // fused GAT stage -> pooled (NG groups per block)
  gat_fused_kernel<<<G / NG, 256, 0, stream>>>(X, g1Wl, g1bl, g1Wr, g1br, g1at, g1bi,
                                               Wl2t, g2bl, Wr2t, g2br, g2at, g2bi,
                                               pooled);

  // input projection + positional encoding
  build_tin_kernel<<<(M * 288 + 255) / 256, 256, 0, stream>>>(pooled, XS, Ain, M);
  gemm_wmma_kernel<<<M / 64, 256, 0, stream>>>(Ain, Wint, bin, t, nullptr,
                                               124, 8, 288, 0);
  pe_kernel<<<(M * 128 + 255) / 256, 256, 0, stream>>>(t, th, M);

  for (int l = 0; l < 2; ++l){
    const int base = 16 + 16 * l;
    const float* bq   = (const float*)d_in[base + 1];
    const float* bk   = (const float*)d_in[base + 3];
    const float* bv   = (const float*)d_in[base + 5];
    const float* bo   = (const float*)d_in[base + 7];
    const float* b1   = (const float*)d_in[base + 9];
    const float* b2   = (const float*)d_in[base +11];
    const float* ln1g = (const float*)d_in[base +12];
    const float* ln1b = (const float*)d_in[base +13];
    const float* ln2g = (const float*)d_in[base +14];
    const float* ln2b = (const float*)d_in[base +15];

    gemm_wmma_kernel<<<M / 64, 256, 0, stream>>>(th, Wqt[l], bq, nullptr, qh, 124, 8, 128, 0);
    gemm_wmma_kernel<<<M / 64, 256, 0, stream>>>(th, Wkt[l], bk, nullptr, kh, 124, 8, 128, 0);
    gemm_wmma_kernel<<<M / 64, 256, 0, stream>>>(th, Wvt[l], bv, nullptr, vh, 124, 8, 128, 0);
    attn_kernel<<<256 * 4, 128, 0, stream>>>(qh, kh, vh, oh);
    gemm_wmma_kernel<<<M / 64, 256, 0, stream>>>(oh, Wot[l], bo, gbuf, nullptr, 124, 8, 128, 0);
    ln_kernel<<<(M + 255) / 256, 256, 0, stream>>>(gbuf, t, ln1g, ln1b, t2, t2h, M);
    gemm_wmma_kernel<<<M / 64, 256, 0, stream>>>(t2h, W1t[l], b1, nullptr, ffh, 64, 4, 128, 1);
    gemm_wmma_kernel<<<M / 64, 256, 0, stream>>>(ffh, W2t[l], b2, gbuf, nullptr, 124, 8, 64, 0);
    ln_kernel<<<(M + 255) / 256, 256, 0, stream>>>(gbuf, t2, ln2g, ln2b, t, th, M);
  }

  final_kernel<<<256, 128, 0, stream>>>(t, Wout, bout, (float*)d_out);
}
